// LocalFeatureAggregation_6030134083769
// MI455X (gfx1250) — matrix operand order repaired and checked by reading, not compiled, not run on hardware
//
#include <hip/hip_runtime.h>

#define Bb    8
#define Nn    4096
#define Kk    16
#define CIN   64
#define CMID  64
#define COUT  128
#define CCAT  131
#define EPSf  1e-5f
#define WAVES 8

typedef __attribute__((ext_vector_type(16))) _Float16 v16h;
typedef __attribute__((ext_vector_type(8)))  float    v8f;

// ---------------------------------------------------------------------------
// Kernel 1: brute-force KNN (k=16) per point, candidates tiled through LDS,
// register-resident sorted top-16 (fully unrolled insertion, no scratch).
// ---------------------------------------------------------------------------
__global__ __launch_bounds__(256) void knn_kernel(const float* __restrict__ xyz,
                                                  int* __restrict__ idx_out) {
  __shared__ float sx[256], sy[256], sz[256];
  const int b = blockIdx.x >> 4;                       // 16 blocks per batch
  const int n = ((blockIdx.x & 15) << 8) | threadIdx.x;
  const float* base = xyz + (size_t)b * Nn * 3;
  const float qx = base[n * 3 + 0];
  const float qy = base[n * 3 + 1];
  const float qz = base[n * 3 + 2];
  float bd[Kk];
  int   bi[Kk];
#pragma unroll
  for (int j = 0; j < Kk; ++j) { bd[j] = 3.4e38f; bi[j] = 0; }

  for (int t0 = 0; t0 < Nn; t0 += 256) {
    __syncthreads();
    const int c = t0 + threadIdx.x;
    sx[threadIdx.x] = base[c * 3 + 0];
    sy[threadIdx.x] = base[c * 3 + 1];
    sz[threadIdx.x] = base[c * 3 + 2];
    __syncthreads();
    for (int j = 0; j < 256; ++j) {
      const float dx = qx - sx[j];
      const float dy = qy - sy[j];
      const float dz = qz - sz[j];
      const float d  = dx * dx + dy * dy + dz * dz;
      if (d < bd[Kk - 1]) {                 // strict < keeps earlier idx on ties
        bd[Kk - 1] = d; bi[Kk - 1] = t0 + j;
#pragma unroll
        for (int s = Kk - 1; s > 0; --s) {
          if (bd[s] < bd[s - 1]) {
            float td = bd[s]; bd[s] = bd[s - 1]; bd[s - 1] = td;
            int   ti = bi[s]; bi[s] = bi[s - 1]; bi[s - 1] = ti;
          }
        }
      }
    }
  }
  int* o = idx_out + (size_t)(b * Nn + n) * Kk;
#pragma unroll
  for (int j = 0; j < Kk; ++j) o[j] = bi[j];
}

// ---------------------------------------------------------------------------
// helpers for register-side A-fragment construction
// ---------------------------------------------------------------------------
__device__ __forceinline__ v16h pack4(float4 a, float4 b, float4 c, float4 d) {
  v16h rr;
  rr[0]  = (_Float16)a.x; rr[1]  = (_Float16)a.y;
  rr[2]  = (_Float16)a.z; rr[3]  = (_Float16)a.w;
  rr[4]  = (_Float16)b.x; rr[5]  = (_Float16)b.y;
  rr[6]  = (_Float16)b.z; rr[7]  = (_Float16)b.w;
  rr[8]  = (_Float16)c.x; rr[9]  = (_Float16)c.y;
  rr[10] = (_Float16)c.z; rr[11] = (_Float16)c.w;
  rr[12] = (_Float16)d.x; rr[13] = (_Float16)d.y;
  rr[14] = (_Float16)d.z; rr[15] = (_Float16)d.w;
  return rr;
}
__device__ __forceinline__ float4 sub4(float4 a, float4 b) {
  float4 rr; rr.x = a.x - b.x; rr.y = a.y - b.y; rr.z = a.z - b.z; rr.w = a.w - b.w;
  return rr;
}

// ---------------------------------------------------------------------------
// Kernel 2: fused gather + 3x (1x1conv + folded BN + ReLU) + K-max, WMMA.
// - Weights pre-permuted in LDS to per-lane fragment order (v16h loads =
//   2x ds_load_b128 each).
// - Layer-1 bias embedded as K-row 131 of B1 (A supplies 1.0 there), so
//   accumulators start from the WMMA inline-0 C operand.
// - Layer-1 A built in registers straight from global float4s (L2-resident).
// - h1/h2 exchanged via a packed per-wave LDS buffer in A-fragment order.
// ---------------------------------------------------------------------------
__global__ __launch_bounds__(256) void lfa_mlp_kernel(
    const float* __restrict__ xyz, const float* __restrict__ features,
    const int* __restrict__ knn,
    const float* __restrict__ W1, const float* __restrict__ g1,
    const float* __restrict__ b1, const float* __restrict__ m1,
    const float* __restrict__ v1,
    const float* __restrict__ W2, const float* __restrict__ g2,
    const float* __restrict__ b2, const float* __restrict__ m2,
    const float* __restrict__ v2,
    const float* __restrict__ W3, const float* __restrict__ g3,
    const float* __restrict__ b3, const float* __restrict__ m3,
    const float* __restrict__ v3,
    float* __restrict__ out) {
  // ---- static LDS: 61,440 bytes ----
  __shared__ _Float16 sB1p[5 * 4 * 32 * 16];   // 20480 B  fragment-packed
  __shared__ _Float16 sB2p[2 * 4 * 32 * 16];   //  8192 B
  __shared__ _Float16 sB3p[2 * 8 * 32 * 16];   // 16384 B
  __shared__ _Float16 sA[WAVES][2 * 32 * 16];  // 16384 B packed h1/h2

  const int tid  = threadIdx.x;
  const int w    = tid >> 5;
  const int lane = tid & 31;
  const int r    = lane & 15;          // A row / C-D column-within-tile
  const int hi16 = lane & 16;
  const int hi8  = hi16 ? 8 : 0;
  const int mrow = hi16 ? 8 : 0;

  // ---- stage BN-folded f16 weights, permuted to fragment order ----
  // offset = ((kc*T + t)*32 + ln)*16 + h ;  B map: k = kc*32 + ((h&8)<<1) + ((ln&16)>>1) + (h&7),
  // o = t*16 + (ln&15)   (ISA 7.12.2 16-bit B layout).  Row k==131 of B1
  // carries the folded bias (A provides 1.0 at c=131); rows 132..159 zero.
  for (int i = tid; i < 5 * 4 * 32 * 16; i += 256) {
    const int h = i & 15, ln = (i >> 4) & 31, t = (i >> 9) & 3, kc = i >> 11;
    const int k = kc * 32 + ((h & 8) << 1) + ((ln & 16) >> 1) + (h & 7);
    const int o = t * 16 + (ln & 15);
    const float s = g1[o] * rsqrtf(v1[o] + EPSf);
    float wv = 0.f;
    if (k < CCAT)       wv = W1[o * CCAT + k] * s;
    else if (k == CCAT) wv = b1[o] - m1[o] * s;          // bias row
    sB1p[i] = (_Float16)wv;
  }
  for (int i = tid; i < 2 * 4 * 32 * 16; i += 256) {
    const int h = i & 15, ln = (i >> 4) & 31, t = (i >> 9) & 3, kc = i >> 11;
    const int k = kc * 32 + ((h & 8) << 1) + ((ln & 16) >> 1) + (h & 7);
    const int o = t * 16 + (ln & 15);
    sB2p[i] = (_Float16)(W2[o * CMID + k] * (g2[o] * rsqrtf(v2[o] + EPSf)));
  }
  for (int i = tid; i < 2 * 8 * 32 * 16; i += 256) {
    const int h = i & 15, ln = (i >> 4) & 31, t = (i >> 9) & 7, kc = i >> 12;
    const int k = kc * 32 + ((h & 8) << 1) + ((ln & 16) >> 1) + (h & 7);
    const int o = t * 16 + (ln & 15);
    sB3p[i] = (_Float16)(W3[o * CMID + k] * (g3[o] * rsqrtf(v3[o] + EPSf)));
  }

  // ---- per-lane folded biases for layers 2/3, hoisted out of the loop ----
  float bv2[4], bv3[8];
#pragma unroll
  for (int t = 0; t < 4; ++t) {
    const int o = t * 16 + r;
    const float s = g2[o] * rsqrtf(v2[o] + EPSf);
    bv2[t] = b2[o] - m2[o] * s;
  }
#pragma unroll
  for (int t = 0; t < 8; ++t) {
    const int o = t * 16 + r;
    const float s = g3[o] * rsqrtf(v3[o] + EPSf);
    bv3[t] = b3[o] - m3[o] * s;
  }
  __syncthreads();

  const int totalWaves = gridDim.x * WAVES;
  const int gw = blockIdx.x * WAVES + w;
  const int beta16 = (r & 8) ? 16 : 0;
  const int hb     = r & 7;

  for (int p = gw; p < Bb * Nn; p += totalWaves) {  // uniform 4 iters per wave
    const int b = p >> 12;
    const int n = p & (Nn - 1);
    const float* fb   = features + (size_t)b * Nn * CIN;
    const float* ctrF = fb + (size_t)n * CIN;
    const int g = knn[(size_t)p * Kk + r];          // this lane's neighbor row
    const float* nbrF = fb + (size_t)g * CIN;

    // ---- layer-1 A in registers: lane touches channels {hi8..hi8+7,
    //      16+hi8..23+hi8} per 32-chunk (ISA 16-bit A layout).  Center
    //      float4s serve both feat_rep chunks (0,1) and feat_c (2,3). ----
    const float4 cA0 = *(const float4*)(ctrF + hi8);
    const float4 cA1 = *(const float4*)(ctrF + hi8 + 4);
    const float4 cB0 = *(const float4*)(ctrF + 16 + hi8);
    const float4 cB1 = *(const float4*)(ctrF + 20 + hi8);
    const float4 cC0 = *(const float4*)(ctrF + 32 + hi8);
    const float4 cC1 = *(const float4*)(ctrF + 36 + hi8);
    const float4 cD0 = *(const float4*)(ctrF + 48 + hi8);
    const float4 cD1 = *(const float4*)(ctrF + 52 + hi8);
    const float4 nA0 = *(const float4*)(nbrF + hi8);
    const float4 nA1 = *(const float4*)(nbrF + hi8 + 4);
    const float4 nB0 = *(const float4*)(nbrF + 16 + hi8);
    const float4 nB1 = *(const float4*)(nbrF + 20 + hi8);
    const float4 nC0 = *(const float4*)(nbrF + 32 + hi8);
    const float4 nC1 = *(const float4*)(nbrF + 36 + hi8);
    const float4 nD0 = *(const float4*)(nbrF + 48 + hi8);
    const float4 nD1 = *(const float4*)(nbrF + 52 + hi8);

    v16h A1c[5];
    A1c[0] = pack4(cA0, cA1, cB0, cB1);                       // feat_rep 0..31
    A1c[1] = pack4(cC0, cC1, cD0, cD1);                       // feat_rep 32..63
    A1c[2] = pack4(sub4(nA0, cA0), sub4(nA1, cA1),
                   sub4(nB0, cB0), sub4(nB1, cB1));           // feat_c 64..95
    A1c[3] = pack4(sub4(nC0, cC0), sub4(nC1, cC1),
                   sub4(nD0, cD0), sub4(nD1, cD1));           // feat_c 96..127
    {
      const float* xb = xyz + (size_t)b * Nn * 3;
      const float dx = xb[g * 3 + 0] - xb[n * 3 + 0];
      const float dy = xb[g * 3 + 1] - xb[n * 3 + 1];
      const float dz = xb[g * 3 + 2] - xb[n * 3 + 2];
      v16h a4;
#pragma unroll
      for (int h = 0; h < 16; ++h) a4[h] = (_Float16)0.f;
      if (!hi16) {                 // c=128..131 live only in lanes 0..15 (h<4)
        a4[0] = (_Float16)dx; a4[1] = (_Float16)dy; a4[2] = (_Float16)dz;
        a4[3] = (_Float16)1.f;     // c=131: multiplies the bias row of B1
      }
      A1c[4] = a4;
    }

    // ---- layer 1: 16x160 @ 160x64 (bias folded into K-row 131) ----
    v8f acc1[4] = {};              // lowers to WMMA inline SRC2 = 0
#pragma unroll
    for (int kc = 0; kc < 5; ++kc) {
      const v16h a = A1c[kc];
      v16h bf[4];
#pragma unroll
      for (int t = 0; t < 4; ++t)
        bf[t] = *(const v16h*)&sB1p[((kc * 4 + t) * 32 + lane) * 16];
#pragma unroll
      for (int t = 0; t < 4; ++t)
        acc1[t] = __builtin_amdgcn_wmma_f32_16x16x32_f16(
            false, a, false, bf[t], (short)0, acc1[t], false, false);
    }

    // ---- ReLU(h1) -> sA, written directly in layer-2 A-fragment order ----
    // value (M, ch): dest = ((ch>>5)*32 + M + 16*((ch>>3)&1))*16
    //                      + ((ch>>4)&1)*8 + (ch&7)
#pragma unroll
    for (int t = 0; t < 4; ++t) {
      const int kc2 = t >> 1;
      const int hh  = ((t & 1) << 3) + hb;
#pragma unroll
      for (int q = 0; q < 8; ++q) {
        const float vv = fmaxf(acc1[t][q], 0.f);
        sA[w][(kc2 * 32 + q + mrow + beta16) * 16 + hh] = (_Float16)vv;
      }
    }
    asm volatile("s_wait_dscnt 0" ::: "memory");     // wave-local LDS fence

    // ---- layer 2: 16x64 @ 64x64 ----
    v8f acc2[4] = {};
#pragma unroll
    for (int kc = 0; kc < 2; ++kc) {
      const v16h a = *(const v16h*)&sA[w][(kc * 32 + lane) * 16];
      v16h bf[4];
#pragma unroll
      for (int t = 0; t < 4; ++t)
        bf[t] = *(const v16h*)&sB2p[((kc * 4 + t) * 32 + lane) * 16];
#pragma unroll
      for (int t = 0; t < 4; ++t)
        acc2[t] = __builtin_amdgcn_wmma_f32_16x16x32_f16(
            false, a, false, bf[t], (short)0, acc2[t], false, false);
    }
#pragma unroll
    for (int t = 0; t < 4; ++t) {                    // bias + ReLU -> sA
      const int kc2 = t >> 1;
      const int hh  = ((t & 1) << 3) + hb;
#pragma unroll
      for (int q = 0; q < 8; ++q) {
        const float vv = fmaxf(acc2[t][q] + bv2[t], 0.f);
        sA[w][(kc2 * 32 + q + mrow + beta16) * 16 + hh] = (_Float16)vv;
      }
    }
    asm volatile("s_wait_dscnt 0" ::: "memory");

    // ---- layer 3: 16x64 @ 64x128, then bias + ReLU + max over neighbors ----
    v8f acc3[8] = {};
#pragma unroll
    for (int kc = 0; kc < 2; ++kc) {
      const v16h a = *(const v16h*)&sA[w][(kc * 32 + lane) * 16];
      v16h bf[8];
#pragma unroll
      for (int t = 0; t < 8; ++t)
        bf[t] = *(const v16h*)&sB3p[((kc * 8 + t) * 32 + lane) * 16];
#pragma unroll
      for (int t = 0; t < 8; ++t)
        acc3[t] = __builtin_amdgcn_wmma_f32_16x16x32_f16(
            false, a, false, bf[t], (short)0, acc3[t], false, false);
    }
    float* orow = out + (size_t)p * COUT;
#pragma unroll
    for (int t = 0; t < 8; ++t) {
      float mx = acc3[t][0];
#pragma unroll
      for (int q = 1; q < 8; ++q) mx = fmaxf(mx, acc3[t][q]);
      mx = fmaxf(mx, __shfl_xor(mx, 16, 32));        // merge the two M-halves
      mx = fmaxf(mx + bv3[t], 0.f);                  // bias once, ReLU floor
      if (!hi16) orow[t * 16 + r] = mx;              // lanes 0..15: 64B store
    }
  }
}

// ---------------------------------------------------------------------------
extern "C" void kernel_launch(void* const* d_in, const int* in_sizes, int n_in,
                              void* d_out, int out_size, void* d_ws, size_t ws_size,
                              hipStream_t stream) {
  const float* xyz      = (const float*)d_in[0];
  const float* features = (const float*)d_in[1];
  const float* W1 = (const float*)d_in[2];
  const float* g1 = (const float*)d_in[3];
  const float* b1 = (const float*)d_in[4];
  const float* m1 = (const float*)d_in[5];
  const float* v1 = (const float*)d_in[6];
  const float* W2 = (const float*)d_in[7];
  const float* g2 = (const float*)d_in[8];
  const float* b2 = (const float*)d_in[9];
  const float* m2 = (const float*)d_in[10];
  const float* v2 = (const float*)d_in[11];
  const float* W3 = (const float*)d_in[12];
  const float* g3 = (const float*)d_in[13];
  const float* b3 = (const float*)d_in[14];
  const float* m3 = (const float*)d_in[15];
  const float* v3 = (const float*)d_in[16];
  float* out = (float*)d_out;
  int* idx = (int*)d_ws;                      // B*N*K*4 = 2 MB scratch

  knn_kernel<<<(Bb * Nn) / 256, 256, 0, stream>>>(xyz, idx);
  lfa_mlp_kernel<<<1024, 256, 0, stream>>>(
      xyz, features, idx,
      W1, g1, b1, m1, v1,
      W2, g2, b2, m2, v2,
      W3, g3, b3, m3, v3, out);
}
